// SimpleBiLSTM_4011499454657
// MI455X (gfx1250) — matrix operand order repaired and checked by reading, not compiled, not run on hardware
//
#include <hip/hip_runtime.h>
#include <hip/hip_bf16.h>

#define NSEQ 16384
#define DIN  1024
#define HID  512
#define G4   2048            // 4*H gate rows
#define NOUT 1024
#define SEGS 16              // workgroups per direction in the scan
#define HSEG (HID / SEGS)    // 32 hidden units per segment
#define ROWS (4 * HSEG)      // 128 gate rows per segment
#define NTPW 4               // N tiles per wave in the GEMM (register blocking)

typedef __bf16 v16bf __attribute__((ext_vector_type(16)));
typedef __bf16 v8bf  __attribute__((ext_vector_type(8)));
typedef float  v8f   __attribute__((ext_vector_type(8)));

__device__ __forceinline__ float sigm(float x) { return 1.0f / (1.0f + __expf(-x)); }
__device__ __forceinline__ float tanh_fast(float x) {
    float e = __expf(-2.0f * x);
    return (1.0f - e) / (1.0f + e);
}

// ---------------------------------------------------------------------------
// Init: zero the inter-WG barrier counters, seed the h-exchange buffer.
// ---------------------------------------------------------------------------
__global__ void bilstm_init(const float* __restrict__ h0f, const float* __restrict__ h0b,
                            float* hg, unsigned int* ctr) {
    int t = threadIdx.x;
    if (t < 2) ctr[t] = 0u;
    for (int i = t; i < HID; i += blockDim.x) {
        hg[i]       = h0f[i];
        hg[HID + i] = h0b[i];
    }
}

// ---------------------------------------------------------------------------
// pre = xs @ Wih.T + (bih + bhh), both directions.  bf16 WMMA, fp32 accum.
// One wave owns a 16x64 C strip: the A fragment (and its bf16 conversion) is
// reused across 4 v_wmma_f32_16x16x32_bf16 issues per K step.
// ---------------------------------------------------------------------------
__global__ __launch_bounds__(128) void pre_gemm(
    const float* __restrict__ xs,
    const float* __restrict__ WihF, const float* __restrict__ WihB,
    const float* __restrict__ bihF, const float* __restrict__ bhhF,
    const float* __restrict__ bihB, const float* __restrict__ bhhB,
    float* __restrict__ preF, float* __restrict__ preB) {
    const int lane  = threadIdx.x & 31;
    const int wave  = threadIdx.x >> 5;
    const int tileM = blockIdx.x * 16;
    const int tileN0 = (blockIdx.y * 4 + wave) * (16 * NTPW);   // 64 cols per wave
    const int dir   = blockIdx.z;

    const float* W  = dir ? WihB : WihF;
    const float* bi = dir ? bihB : bihF;
    const float* bh = dir ? bhhB : bhhF;
    float*      pre = dir ? preB : preF;

    const int lh    = lane >> 4;       // lane-half selector
    const int l15   = lane & 15;
    const int arow  = tileM + l15;     // A: M across lanes
    const int akoff = lh * 8;          // A per-half K sub-offset
    const int bkoff = lh * 8;          // B per-half K sub-offset

    const float* aptr = xs + (size_t)arow * DIN;
    const float* bptr[NTPW];
#pragma unroll
    for (int j = 0; j < NTPW; ++j)
        bptr[j] = W + (size_t)(tileN0 + j * 16 + l15) * DIN;

    v8f acc[NTPW] = {};
    for (int k0 = 0; k0 < DIN; k0 += 32) {
        // ---- A fragment: load + convert once, reuse for 4 WMMAs ----
        const float* ar = aptr + k0;
        float4 a0 = *(const float4*)(ar + akoff);
        float4 a1 = *(const float4*)(ar + akoff + 4);
        float4 a2 = *(const float4*)(ar + 16 + akoff);
        float4 a3 = *(const float4*)(ar + 16 + akoff + 4);
        v16bf a;
        a[0]  = (__bf16)a0.x; a[1]  = (__bf16)a0.y; a[2]  = (__bf16)a0.z; a[3]  = (__bf16)a0.w;
        a[4]  = (__bf16)a1.x; a[5]  = (__bf16)a1.y; a[6]  = (__bf16)a1.z; a[7]  = (__bf16)a1.w;
        a[8]  = (__bf16)a2.x; a[9]  = (__bf16)a2.y; a[10] = (__bf16)a2.z; a[11] = (__bf16)a2.w;
        a[12] = (__bf16)a3.x; a[13] = (__bf16)a3.y; a[14] = (__bf16)a3.z; a[15] = (__bf16)a3.w;

#pragma unroll
        for (int j = 0; j < NTPW; ++j) {
            const float* br = bptr[j] + k0 + bkoff;
            float4 b0 = *(const float4*)(br);
            float4 b1 = *(const float4*)(br + 4);
            float4 b2 = *(const float4*)(br + 16);
            float4 b3 = *(const float4*)(br + 20);
            v16bf b;
            b[0]  = (__bf16)b0.x; b[1]  = (__bf16)b0.y; b[2]  = (__bf16)b0.z; b[3]  = (__bf16)b0.w;
            b[4]  = (__bf16)b1.x; b[5]  = (__bf16)b1.y; b[6]  = (__bf16)b1.z; b[7]  = (__bf16)b1.w;
            b[8]  = (__bf16)b2.x; b[9]  = (__bf16)b2.y; b[10] = (__bf16)b2.z; b[11] = (__bf16)b2.w;
            b[12] = (__bf16)b3.x; b[13] = (__bf16)b3.y; b[14] = (__bf16)b3.z; b[15] = (__bf16)b3.w;

            acc[j] = __builtin_amdgcn_wmma_f32_16x16x32_bf16(false, a, false, b, (short)0,
                                                             acc[j], false, false);
        }
    }

#pragma unroll
    for (int j = 0; j < NTPW; ++j) {
        const int bcol = tileN0 + j * 16 + l15;
        const float bias = bi[bcol] + bh[bcol];
#pragma unroll
        for (int r = 0; r < 8; ++r) {
            const int m = tileM + r + lh * 8;           // C layout: VGPR r, lane-half -> M
            pre[(size_t)m * G4 + bcol] = acc[j][r] + bias;
        }
    }
}

// ---------------------------------------------------------------------------
// Sequential scan.  32 workgroups: 16 per direction, each owns 32 hidden
// units (128 gate rows of Whh staged to LDS as bf16 = 128 KB).  Per step:
// LDS matvec with mixed-precision FMAs + wave32 pair reduction + pointwise
// LSTM update, then an inter-workgroup barrier (hardware cluster barrier
// when dispatched as a cluster, agent-scope atomic counter otherwise).
// ---------------------------------------------------------------------------
__global__ __launch_bounds__(256) void bilstm_scan(
    const float* __restrict__ preF, const float* __restrict__ preB,
    const float* __restrict__ WhhF, const float* __restrict__ WhhB,
    const float* __restrict__ c0F,  const float* __restrict__ c0B,
    float* hg, unsigned int* ctr, float* __restrict__ out) {
    __shared__ __bf16 whh_s[ROWS * HID];    // 128 rows x 512 bf16 = 128 KB
    __shared__ float hcur[HID];
    __shared__ float gbuf[4][HSEG];
    __shared__ float c_s[HSEG];

    const int tid  = threadIdx.x;
    const int blk  = blockIdx.x;
    const int dir  = blk >> 4;
    const int seg  = blk & (SEGS - 1);
    const int hid0 = seg * HSEG;

    const float* Whh = dir ? WhhB : WhhF;
    const float* pre = dir ? preB : preF;
    const float* c0  = dir ? c0B  : c0F;

    // Stage this segment's Whh rows into LDS (fp32 -> bf16, RNE convert).
    for (int idx = tid; idx < ROWS * HID; idx += 256) {
        const int r    = idx >> 9;          // 0..127
        const int k    = idx & 511;
        const int gate = r >> 5;
        const int nl   = r & 31;
        whh_s[idx] = (__bf16)Whh[(size_t)(gate * HID + hid0 + nl) * HID + k];
    }
    if (tid < HSEG) c_s[tid] = c0[hid0 + tid];

    volatile float* hgv = hg;
    for (int i = tid; i < HID; i += 256) hcur[i] = hgv[dir * HID + i];
    __syncthreads();

    const int row  = tid >> 1;   // 0..127 gate rows
    const int half = tid & 1;    // each thread sums half a row (256 terms)
    const v8bf* wrow8 = (const v8bf*)&whh_s[row * HID + half * 256];
    const float* hp = &hcur[half * 256];
    const int gate = row >> 5;
    const int nl   = row & 31;
    const int use_cluster = (__builtin_amdgcn_cluster_id_x() != 0);

    for (int t = 0; t < NSEQ; ++t) {
        const int tg = dir ? (NSEQ - 1 - t) : t;

        // Prefetch next timestep's pre row segment (global_prefetch_b8).
        const int tnext = dir ? (tg - 1) : (tg + 1);
        if ((unsigned)tnext < (unsigned)NSEQ && half == 0)
            __builtin_prefetch(pre + (size_t)tnext * G4 + gate * HID + hid0 + nl, 0, 1);

        float sum = 0.f;
#pragma unroll 4
        for (int kk = 0; kk < 32; ++kk) {          // 32 x 8 = 256 terms
            v8bf  w  = wrow8[kk];
            float4 h0 = *(const float4*)(hp + kk * 8);
            float4 h1 = *(const float4*)(hp + kk * 8 + 4);
            sum = fmaf((float)w[0], h0.x, sum);
            sum = fmaf((float)w[1], h0.y, sum);
            sum = fmaf((float)w[2], h0.z, sum);
            sum = fmaf((float)w[3], h0.w, sum);
            sum = fmaf((float)w[4], h1.x, sum);
            sum = fmaf((float)w[5], h1.y, sum);
            sum = fmaf((float)w[6], h1.z, sum);
            sum = fmaf((float)w[7], h1.w, sum);
        }
        sum += __shfl_xor(sum, 1, 32);  // combine the two halves (wave32)

        if (half == 0)
            gbuf[gate][nl] = pre[(size_t)tg * G4 + gate * HID + hid0 + nl] + sum;
        __syncthreads();

        if (tid < HSEG) {
            const float ig = sigm(gbuf[0][tid]);
            const float fg = sigm(gbuf[1][tid]);
            const float gg = tanh_fast(gbuf[2][tid]);
            const float og = sigm(gbuf[3][tid]);
            const float c  = fg * c_s[tid] + ig * gg;
            c_s[tid]       = c;
            const float h  = og * tanh_fast(c);
            out[(size_t)tg * NOUT + dir * HID + hid0 + tid] = h;
            hgv[dir * HID + hid0 + tid] = h;
        }
        __threadfence();
        __syncthreads();

        // Inter-workgroup barrier among the 16 segments of this direction.
        if (use_cluster) {
            __builtin_amdgcn_s_cluster_barrier();   // s_barrier_signal/wait -3
        } else {
            if (tid == 0) {
                __hip_atomic_fetch_add(&ctr[dir], 1u, __ATOMIC_ACQ_REL,
                                       __HIP_MEMORY_SCOPE_AGENT);
                const unsigned int target = (unsigned int)(SEGS * (t + 1));
                while (__hip_atomic_load(&ctr[dir], __ATOMIC_ACQUIRE,
                                         __HIP_MEMORY_SCOPE_AGENT) < target)
                    __builtin_amdgcn_s_sleep(2);
            }
            __syncthreads();
        }

        // Re-broadcast the full h vector for the next step.
        for (int i2 = tid; i2 < HID; i2 += 256) hcur[i2] = hgv[dir * HID + i2];
        __syncthreads();
    }
}

// ---------------------------------------------------------------------------
extern "C" void kernel_launch(void* const* d_in, const int* in_sizes, int n_in,
                              void* d_out, int out_size, void* d_ws, size_t ws_size,
                              hipStream_t stream) {
    const float* xs   = (const float*)d_in[0];
    const float* WihF = (const float*)d_in[1];
    const float* WhhF = (const float*)d_in[2];
    const float* bihF = (const float*)d_in[3];
    const float* bhhF = (const float*)d_in[4];
    const float* h0F  = (const float*)d_in[5];
    const float* c0F  = (const float*)d_in[6];
    const float* WihB = (const float*)d_in[7];
    const float* WhhB = (const float*)d_in[8];
    const float* bihB = (const float*)d_in[9];
    const float* bhhB = (const float*)d_in[10];
    const float* h0B  = (const float*)d_in[11];
    const float* c0B  = (const float*)d_in[12];
    float* out = (float*)d_out;

    char* ws = (char*)d_ws;
    const size_t preBytes = (size_t)NSEQ * G4 * sizeof(float);  // 128 MB each
    float*        preF = (float*)(ws);
    float*        preB = (float*)(ws + preBytes);
    float*        hg   = (float*)(ws + 2 * preBytes);
    unsigned int* ctr  = (unsigned int*)(ws + 2 * preBytes + 2 * HID * sizeof(float));

    bilstm_init<<<1, 256, 0, stream>>>(h0F, h0B, hg, ctr);
    // grid: M tiles x (2048 / (4 waves * 64 cols)) x directions
    pre_gemm<<<dim3(NSEQ / 16, G4 / (4 * 16 * NTPW), 2), 128, 0, stream>>>(
        xs, WihF, WihB, bihF, bhhF, bihB, bhhB, preF, preB);
    bilstm_scan<<<dim3(2 * SEGS), 256, 0, stream>>>(
        preF, preB, WhhF, WhhB, c0F, c0B, hg, ctr, out);
}
